// Block_88639535055160
// MI455X (gfx1250) — matrix operand order repaired and checked by reading, not compiled, not run on hardware
//
#include <hip/hip_runtime.h>

// ---------------------------------------------------------------------------
// Types
// ---------------------------------------------------------------------------
typedef __bf16 bf16;
typedef __attribute__((ext_vector_type(16))) __bf16 v16bf;
typedef __attribute__((ext_vector_type(8)))  __bf16 v8bf;
typedef __attribute__((ext_vector_type(8)))  float  v8f;

#define BB 2
#define TT 2048
#define CC 1024
#define HH 16
#define DD 64
#define MM (BB * TT)   // 4096 rows
#define FF (4 * CC)    // 4096 ffn dim

// ---------------------------------------------------------------------------
// Helpers
// ---------------------------------------------------------------------------
__device__ __forceinline__ float lane_bcast(float v, int src_lane) {
  int r = __builtin_amdgcn_ds_bpermute(src_lane << 2, __builtin_bit_cast(int, v));
  return __builtin_bit_cast(float, r);
}

// A-fragment (16xK-chunk, 16-bit): lane holds row (lane&15); half-wave K split
// of 8 within each 16-group (ISA 7.12.2).  p must point at A[row*lda + k0 + kb]
// where kb = 8*(lane>>4).  Elements 0..7 <- p[0..7], 8..15 <- p[16..23].
__device__ __forceinline__ v16bf load_fragA(const bf16* p) {
  union { v16bf v; v8bf h[2]; } u;
  u.h[0] = *(const v8bf*)(p);
  u.h[1] = *(const v8bf*)(p + 16);
  return u.v;
}

// B-fragment (Kx16, 16-bit): lane holds column (lane&15); same K order as A:
// elements 0..7 <- p[0..7], 8..15 <- p[16..23].
// p must point at Bt[col*ldb + k0 + 8*(lane>>4)]  (Bt stored [N,K]).
__device__ __forceinline__ v16bf load_fragB(const bf16* p) {
  union { v16bf v; v8bf h[2]; } u;
  u.h[0] = *(const v8bf*)(p);
  u.h[1] = *(const v8bf*)(p + 16);
  return u.v;
}

__device__ __forceinline__ v8f wmma_bf16(v16bf a, v16bf b, v8f c) {
  return __builtin_amdgcn_wmma_f32_16x16x32_bf16(false, a, false, b,
                                                 (short)0, c, false, false);
}

// ---------------------------------------------------------------------------
// Weight packing: fp32 -> bf16, laid out [N, K] (row n contiguous in K)
// ---------------------------------------------------------------------------
__global__ __launch_bounds__(256) void pack_qkv(const float* __restrict__ W,
                                                bf16* __restrict__ out) {
  // W: [H, C, D] fp32.  out[n*C + c] with n = h*D + d.
  int idx = blockIdx.x * 256 + threadIdx.x;     // over CC*CC
  int n = idx >> 10, c = idx & 1023;
  int h = n >> 6, d = n & 63;
  out[idx] = (bf16)W[((long)h * CC + c) * DD + d];
}

__global__ __launch_bounds__(256) void pack_tr(const float* __restrict__ W,
                                               bf16* __restrict__ out,
                                               int Kdim, int Ndim) {
  // W: [Kdim, Ndim] fp32  ->  out: [Ndim, Kdim] bf16
  int idx = blockIdx.x * 256 + threadIdx.x;     // over Ndim*Kdim
  int n = idx / Kdim, k = idx - n * Kdim;
  out[idx] = (bf16)W[(long)k * Ndim + n];
}

// ---------------------------------------------------------------------------
// LayerNorm (block per row of 1024) -> bf16
// ---------------------------------------------------------------------------
__global__ __launch_bounds__(256) void ln_kernel(const float* __restrict__ x,
                                                 const float* __restrict__ g,
                                                 const float* __restrict__ bta,
                                                 bf16* __restrict__ out) {
  __shared__ float s1[256];
  __shared__ float s2[256];
  const int row = blockIdx.x;
  const int tid = threadIdx.x;
  const float* xp = x + (long)row * CC;
  float4 v = ((const float4*)xp)[tid];
  s1[tid] = v.x + v.y + v.z + v.w;
  s2[tid] = v.x * v.x + v.y * v.y + v.z * v.z + v.w * v.w;
  __syncthreads();
  for (int off = 128; off > 0; off >>= 1) {
    if (tid < off) { s1[tid] += s1[tid + off]; s2[tid] += s2[tid + off]; }
    __syncthreads();
  }
  float mean = s1[0] * (1.0f / CC);
  float var  = s2[0] * (1.0f / CC) - mean * mean;
  float rstd = rsqrtf(var + 1e-5f);
  float4 gv = ((const float4*)g)[tid];
  float4 bv = ((const float4*)bta)[tid];
  bf16* op = out + (long)row * CC + tid * 4;
  op[0] = (bf16)((v.x - mean) * rstd * gv.x + bv.x);
  op[1] = (bf16)((v.y - mean) * rstd * gv.y + bv.y);
  op[2] = (bf16)((v.z - mean) * rstd * gv.z + bv.z);
  op[3] = (bf16)((v.w - mean) * rstd * gv.w + bv.w);
}

// ---------------------------------------------------------------------------
// WMMA GEMM: C[M,N] = A[M,K](bf16) * Bt[N,K]^T(bf16) (+bias)(+relu)(+resid)
// block = 256 (8 waves); wave tile = 32x64; block tile = 256(M) x 64(N)
// grid = (N/64, M/256)
// outBT (optional): transposed bf16 store as [B*N, T]: used for V so the
// attention kernel can load V B-fragments as pairs of contiguous 16B vectors.
// ---------------------------------------------------------------------------
__global__ __launch_bounds__(256) void gemm_bf16(
    const bf16* __restrict__ A, const bf16* __restrict__ Bt,
    const float* __restrict__ bias, const float* __restrict__ resid,
    float* __restrict__ outF, bf16* __restrict__ outB,
    bf16* __restrict__ outBT, int N, int K, int relu) {
  const int lane = threadIdx.x & 31;
  const int wave = threadIdx.x >> 5;
  const int hf   = lane >> 4;
  const int l16  = lane & 15;
  const int kbA  = hf * 8;    // A fragment half-wave K offset
  const int kbB  = hf * 8;    // B fragment half-wave K offset
  const int m0   = blockIdx.y * 256 + wave * 32;
  const int n0   = blockIdx.x * 64;

  v8f acc[2][4];
#pragma unroll
  for (int i = 0; i < 2; ++i)
#pragma unroll
    for (int j = 0; j < 4; ++j) {
      float bv = bias ? bias[n0 + j * 16 + l16] : 0.0f;
#pragma unroll
      for (int r = 0; r < 8; ++r) acc[i][j][r] = bv;
    }

  for (int k0 = 0; k0 < K; k0 += 32) {
    v16bf a0 = load_fragA(A + (long)(m0 + l16) * K + k0 + kbA);
    v16bf a1 = load_fragA(A + (long)(m0 + 16 + l16) * K + k0 + kbA);
    v16bf b0 = load_fragB(Bt + (long)(n0 +  0 + l16) * K + k0 + kbB);
    v16bf b1 = load_fragB(Bt + (long)(n0 + 16 + l16) * K + k0 + kbB);
    v16bf b2 = load_fragB(Bt + (long)(n0 + 32 + l16) * K + k0 + kbB);
    v16bf b3 = load_fragB(Bt + (long)(n0 + 48 + l16) * K + k0 + kbB);
    acc[0][0] = wmma_bf16(a0, b0, acc[0][0]);
    acc[0][1] = wmma_bf16(a0, b1, acc[0][1]);
    acc[0][2] = wmma_bf16(a0, b2, acc[0][2]);
    acc[0][3] = wmma_bf16(a0, b3, acc[0][3]);
    acc[1][0] = wmma_bf16(a1, b0, acc[1][0]);
    acc[1][1] = wmma_bf16(a1, b1, acc[1][1]);
    acc[1][2] = wmma_bf16(a1, b2, acc[1][2]);
    acc[1][3] = wmma_bf16(a1, b3, acc[1][3]);
  }

#pragma unroll
  for (int i = 0; i < 2; ++i)
#pragma unroll
    for (int j = 0; j < 4; ++j)
#pragma unroll
      for (int r = 0; r < 8; ++r) {
        long row = m0 + i * 16 + r + 8 * hf;
        long col = n0 + j * 16 + l16;
        float v = acc[i][j][r];
        if (relu) v = v > 0.0f ? v : 0.0f;
        if (resid) v += resid[row * N + col];
        if (outF) outF[row * N + col] = v;
        if (outB) outB[row * N + col] = (bf16)v;
        if (outBT) {
          long bidx = row >> 11;          // row / T  (T = 2048)
          long t    = row & (TT - 1);     // row % T
          outBT[(bidx * N + col) * TT + t] = (bf16)v;
        }
      }
}

// ---------------------------------------------------------------------------
// Causal flash attention.
//   Q/K stored [B*T, C] bf16 with col = h*64+d.
//   V  stored transposed [B*C, T] bf16 (row = b*C + h*64 + d, contiguous in t)
//   O  stored [B*T, C] bf16.
// Wave owns 16 queries x D=64.  Computes S^T = K * Q^T so the softmaxed P
// lands directly in A-fragment layout for the P*V WMMA; all fragment loads
// are pairs of contiguous 16-byte vectors.
// grid = (T/128, H, B); block = 256 (8 waves, 128 queries per block)
// ---------------------------------------------------------------------------
__global__ __launch_bounds__(256) void attn_kernel(
    const bf16* __restrict__ Qm, const bf16* __restrict__ Km,
    const bf16* __restrict__ Vt, bf16* __restrict__ Om) {
  const int lane = threadIdx.x & 31;
  const int wave = threadIdx.x >> 5;
  const int hf   = lane >> 4;
  const int l16  = lane & 15;
  const int kbA  = hf * 8;
  const int kbB  = hf * 8;
  const int b = blockIdx.z, h = blockIdx.y;
  const int q0 = blockIdx.x * 128 + wave * 16;
  const long rowbase = (long)b * TT;
  const int colb = h * DD;
  const float NEG_INF = -__builtin_inff();

  const bf16* Qp = Qm + rowbase * CC + colb;
  const bf16* Kp = Km + rowbase * CC + colb;
  const bf16* Vp = Vt + ((long)b * CC + colb) * TT;   // rows = d, cols = t

  // Q^T fragments (B operand: lane = query column), d-chunks 0..31 and 32..63
  v16bf qt0 = load_fragB(Qp + (long)(q0 + l16) * CC + kbB);
  v16bf qt1 = load_fragB(Qp + (long)(q0 + l16) * CC + 32 + kbB);

  v8f o[4];
#pragma unroll
  for (int j = 0; j < 4; ++j)
#pragma unroll
    for (int r = 0; r < 8; ++r) o[j][r] = 0.0f;

  float m_i = NEG_INF;
  float l_i = 0.0f;
  const float scale = 0.125f;  // 1/sqrt(64)
  const int q_lane = q0 + l16; // query this lane tracks in S^T layout
  const int kend = q0 + 16;

  for (int k0 = 0; k0 < kend; k0 += 32) {
    v8f s[2];
#pragma unroll
    for (int t = 0; t < 2; ++t) {
      const int kt = k0 + t * 16;
      v16bf ka0 = load_fragA(Kp + (long)(kt + l16) * CC + kbA);
      v16bf ka1 = load_fragA(Kp + (long)(kt + l16) * CC + 32 + kbA);
      v8f z;
#pragma unroll
      for (int r = 0; r < 8; ++r) z[r] = 0.0f;
      z = wmma_bf16(ka0, qt0, z);   // S^T = K * Q^T
      z = wmma_bf16(ka1, qt1, z);
#pragma unroll
      for (int r = 0; r < 8; ++r) {
        int key = kt + r + 8 * hf;  // C-layout: VGPR row -> key index
        float sv = z[r] * scale;
        s[t][r] = (key > q_lane) ? NEG_INF : sv;
      }
    }
    // online softmax over the 32-key step (per query = per lane column)
    float smax = NEG_INF;
#pragma unroll
    for (int t = 0; t < 2; ++t)
#pragma unroll
      for (int r = 0; r < 8; ++r) smax = fmaxf(smax, s[t][r]);
    smax = fmaxf(smax, lane_bcast(smax, lane ^ 16));
    float mnew  = fmaxf(m_i, smax);
    float alpha = __expf(m_i - mnew);
    float rsum  = 0.0f;
    v16bf pa;   // P in A-fragment layout: element e -> key (e/8)*16 + 8*hf + e%8
#pragma unroll
    for (int t = 0; t < 2; ++t)
#pragma unroll
      for (int r = 0; r < 8; ++r) {
        float p = __expf(s[t][r] - mnew);
        rsum += p;
        pa[t * 8 + r] = (bf16)p;
      }
    rsum += lane_bcast(rsum, lane ^ 16);
    l_i = l_i * alpha + rsum;
    m_i = mnew;
    // rescale O accumulator (O layout: lane = d column, VGPR row = query)
#pragma unroll
    for (int r = 0; r < 8; ++r) {
      float av = lane_bcast(alpha, r + 8 * hf);
#pragma unroll
      for (int j = 0; j < 4; ++j) o[j][r] *= av;
    }
    // P * V: V B-fragment from transposed V -> two contiguous 16B loads
    // (lane = d column j*16+l16, elements 0..7 = keys k0+8*hf .. +7, 8..15 = keys k0+16+8*hf .. +7)
#pragma unroll
    for (int j = 0; j < 4; ++j) {
      v16bf vbf = load_fragB(Vp + (long)(j * 16 + l16) * TT + k0 + kbB);
      o[j] = wmma_bf16(pa, vbf, o[j]);
    }
  }

  float linv = 1.0f / l_i;
#pragma unroll
  for (int r = 0; r < 8; ++r) {
    float lv = lane_bcast(linv, r + 8 * hf);
    long row = q0 + r + 8 * hf;
    bf16* op = Om + (rowbase + row) * CC + colb + l16;
#pragma unroll
    for (int j = 0; j < 4; ++j) op[j * 16] = (bf16)(o[j][r] * lv);
  }
}

// ---------------------------------------------------------------------------
// Host orchestration
// ---------------------------------------------------------------------------
extern "C" void kernel_launch(void* const* d_in, const int* in_sizes, int n_in,
                              void* d_out, int out_size, void* d_ws, size_t ws_size,
                              hipStream_t stream) {
  (void)in_sizes; (void)n_in; (void)out_size; (void)ws_size;
  const float* x     = (const float*)d_in[0];
  const float* Wq    = (const float*)d_in[1];
  const float* Wk    = (const float*)d_in[2];
  const float* Wv    = (const float*)d_in[3];
  const float* Wproj = (const float*)d_in[4];
  const float* bproj = (const float*)d_in[5];
  const float* W1    = (const float*)d_in[6];
  const float* b1    = (const float*)d_in[7];
  const float* W2    = (const float*)d_in[8];
  const float* b2    = (const float*)d_in[9];
  const float* ln1g  = (const float*)d_in[10];
  const float* ln1b  = (const float*)d_in[11];
  const float* ln2g  = (const float*)d_in[12];
  const float* ln2b  = (const float*)d_in[13];

  // workspace layout (bytes)
  char* p = (char*)d_ws;
  const size_t SZ_W   = (size_t)CC * CC * 2;   // 2 MB  packed [C,C] bf16
  const size_t SZ_WF  = (size_t)CC * FF * 2;   // 8 MB  packed [4C,C]/[C,4C]
  const size_t SZ_ACT = (size_t)MM * CC * 2;   // 8 MB  bf16 activation
  const size_t SZ_F32 = (size_t)MM * CC * 4;   // 16 MB fp32 activation
  const size_t SZ_FF  = (size_t)MM * FF * 2;   // 32 MB bf16 ffn activation

  bf16* wqT = (bf16*)p; p += SZ_W;
  bf16* wkT = (bf16*)p; p += SZ_W;
  bf16* wvT = (bf16*)p; p += SZ_W;
  bf16* wpT = (bf16*)p; p += SZ_W;
  bf16* w1T = (bf16*)p; p += SZ_WF;
  bf16* w2T = (bf16*)p; p += SZ_WF;
  bf16* h1  = (bf16*)p; p += SZ_ACT;           // LN1(x);  reused as att-out
  bf16* qb  = (bf16*)p; p += SZ_ACT;           // Q;       reused as LN2(x1)
  bf16* kb  = (bf16*)p; p += SZ_ACT;
  bf16* vt  = (bf16*)p; p += SZ_ACT;           // V transposed [B*C, T]
  float* x1 = (float*)p; p += SZ_F32;          // residual after attention
  bf16* ff1 = (bf16*)p; p += SZ_FF;
  bf16* att = h1;                              // alias: h1 dead after V GEMM
  bf16* h2  = qb;                              // alias: Q dead after attention

  const dim3 blk(256);
  // ---- pack weights to bf16 [N,K] ----
  pack_qkv<<<(CC * CC) / 256, blk, 0, stream>>>(Wq, wqT);
  pack_qkv<<<(CC * CC) / 256, blk, 0, stream>>>(Wk, wkT);
  pack_qkv<<<(CC * CC) / 256, blk, 0, stream>>>(Wv, wvT);
  pack_tr <<<(CC * CC) / 256, blk, 0, stream>>>(Wproj, wpT, CC, CC);
  pack_tr <<<(CC * FF) / 256, blk, 0, stream>>>(W1, w1T, CC, FF);
  pack_tr <<<(FF * CC) / 256, blk, 0, stream>>>(W2, w2T, FF, CC);

  // ---- LN1 ----
  ln_kernel<<<MM, blk, 0, stream>>>(x, ln1g, ln1b, h1);

  // ---- QKV projections (V written transposed for the attention kernel) ----
  const dim3 gC(CC / 64, MM / 256);
  gemm_bf16<<<gC, blk, 0, stream>>>(h1, wqT, nullptr, nullptr, nullptr, qb, nullptr, CC, CC, 0);
  gemm_bf16<<<gC, blk, 0, stream>>>(h1, wkT, nullptr, nullptr, nullptr, kb, nullptr, CC, CC, 0);
  gemm_bf16<<<gC, blk, 0, stream>>>(h1, wvT, nullptr, nullptr, nullptr, nullptr, vt, CC, CC, 0);

  // ---- causal attention ----
  attn_kernel<<<dim3(TT / 128, HH, BB), blk, 0, stream>>>(qb, kb, vt, att);

  // ---- output projection + residual ----
  gemm_bf16<<<gC, blk, 0, stream>>>(att, wpT, bproj, x, x1, nullptr, nullptr, CC, CC, 0);

  // ---- LN2 ----
  ln_kernel<<<MM, blk, 0, stream>>>(x1, ln2g, ln2b, h2);

  // ---- FFN ----
  const dim3 gF(FF / 64, MM / 256);
  gemm_bf16<<<gF, blk, 0, stream>>>(h2, w1T, b1, nullptr, nullptr, ff1, nullptr, FF, CC, 1);
  gemm_bf16<<<gC, blk, 0, stream>>>(ff1, w2T, b2, x1, (float*)d_out, nullptr, nullptr, CC, FF, 0);
}
